// MixEHR_34454227649001
// MI455X (gfx1250) — matrix operand order repaired and checked by reading, not compiled, not run on hardware
//
#include <hip/hip_runtime.h>

typedef __attribute__((ext_vector_type(16))) __bf16 v16bf;
typedef __attribute__((ext_vector_type(8)))  float  v8f;

#define MINIF 1e-6f
#define Bn 256
#define Vn 4096
#define Kn 64
#define LN2F 0.6931471805599453f

// rho = 1/(it+5)^0.9 = exp2(-0.9*log2(it+5)); argument >= 5, no fixup needed.
static __device__ inline float fast_rho(int it) {
    float x = (float)(it + 5);
    return __builtin_amdgcn_exp2f(-0.9f * __builtin_amdgcn_logf(x));
}

// ---------------- Kernel 1: denom[k] = sum_v (beta[v,k] + exp_n[v,k]) ----------------
__global__ __launch_bounds__(256)
void k_denom(const float* __restrict__ beta, const float* __restrict__ expn,
             float* __restrict__ denom) {
    __shared__ float part[256];
    int t = threadIdx.x;
    int k = t & 63;
    int p = t >> 6;               // 4 partial sums per k
    float s = 0.f;
    for (int v = p; v < Vn; v += 4) {
        int i = v * Kn + k;
        s += beta[i] + expn[i];
    }
    part[t] = s;
    __syncthreads();
    if (t < 64)
        denom[t] = part[t] + part[64 + t] + part[128 + t] + part[192 + t];
}

// ---------------- Kernel 2: phi = (beta+exp_n)/denom ; f32 + bf16 + bf16-transposed ----
__global__ __launch_bounds__(256)
void k_phi(const float* __restrict__ beta, const float* __restrict__ expn,
           const float* __restrict__ denom,
           float* __restrict__ phF, __bf16* __restrict__ phB, __bf16* __restrict__ phT) {
    int i = blockIdx.x * 256 + threadIdx.x;   // i < V*K
    int k = i & 63, v = i >> 6;
    float ph = (beta[i] + expn[i]) / denom[k];
    phF[i] = ph;
    __bf16 h = (__bf16)ph;
    phB[i] = h;
    phT[k * Vn + v] = h;
}

// ---------------- Kernel 3: theta = alpha*pi[idx] + exp_m[idx] -------------------------
__global__ __launch_bounds__(256)
void k_theta(const float* __restrict__ alpha, const float* __restrict__ pi,
             const float* __restrict__ expm, const int* __restrict__ bidx,
             float* __restrict__ thF, __bf16* __restrict__ thB, __bf16* __restrict__ thT) {
    int i = blockIdx.x * 256 + threadIdx.x;   // i < B*K
    int k = i & 63, b = i >> 6;
    int d = bidx[b];
    float th = alpha[k] * pi[d * Kn + k] + expm[d * Kn + k];
    thF[i] = th;
    __bf16 h = (__bf16)th;
    thB[i] = h;
    thT[k * Bn + b] = h;
}

// ---------------- Kernel 4: S = theta . phi^T (WMMA), w = cnt/(S+eps), qz log pass -----
// grid = (V/128, B/16), 8 waves/block; each wave owns a 16x16 S tile.
__global__ __launch_bounds__(256)
void k_gamma(const int* __restrict__ bow,
             const float* __restrict__ thF, const float* __restrict__ phF,
             const __bf16* __restrict__ thB, const __bf16* __restrict__ phB,
             __bf16* __restrict__ W, __bf16* __restrict__ WT,
             float* __restrict__ qzPart) {
    __shared__ float thS[16 * 64];
    __shared__ float phS[128 * 68];    // pad 68 to avoid bank conflicts on row access
    __shared__ float red[256];
    int tid  = threadIdx.x;
    int lane = tid & 31, wave = tid >> 5;
    int v0 = blockIdx.x * 128, b0 = blockIdx.y * 16;

    for (int i = tid; i < 16 * 64; i += 256)
        thS[i] = thF[b0 * 64 + i];
    for (int i = tid; i < 128 * 64; i += 256) {
        int v = i >> 6, k = i & 63;
        phS[v * 68 + k] = phF[(v0 + v) * 64 + k];
    }
    __syncthreads();

    int n = lane & 15, hi = lane >> 4, khalf = hi * 8;
    // WMMA accumulate over K=64 in two bf16 16x16x32 steps
    v8f acc = {};
#pragma unroll
    for (int c = 0; c < 2; ++c) {
        int kofs = c * 32;
        v16bf a, bmat;
        const __bf16* ap0 = thB + (b0 + n) * 64 + kofs + hi * 16;       // A row = lane%16
        const __bf16* ap1 = ap0 + 8;
#pragma unroll
        for (int e = 0; e < 8; ++e) { a[e] = ap0[e]; a[8 + e] = ap1[e]; }
        const __bf16* bp = phB + (v0 + wave * 16 + n) * 64 + kofs + hi * 16;
#pragma unroll
        for (int e = 0; e < 16; ++e) bmat[e] = bp[e];
        acc = __builtin_amdgcn_wmma_f32_16x16x32_bf16(false, a, false, bmat,
                                                      (short)0, acc, false, false);
    }

    int vg = v0 + wave * 16 + n;
    float qacc2 = 0.f;                 // accumulates g*log2(g+MINI); scale by ln2 once
#pragma unroll
    for (int j = 0; j < 8; ++j) {
        int m  = j + 8 * hi;           // C/D row layout
        int bg = b0 + m;
        float S = acc[j];
        int cnt = bow[bg * Vn + vg];
        float w = 0.f;
        if (cnt > 0) {
            // S+MINI >= 1e-6 and well-scaled: raw v_rcp_f32 is sufficient
            float invS = __builtin_amdgcn_rcpf(S + MINIF);
            w = (float)cnt * invS;
            const float* thr = &thS[m * 64];
            const float* phr = &phS[(wave * 16 + n) * 68];
            for (int k = 0; k < 64; ++k) {
                float g = thr[k] * phr[k] * invS;
                // g+MINI >= 1e-6 >> FLT_MIN: raw v_log_f32 (log2), no denorm fixup
                qacc2 += g * __builtin_amdgcn_logf(g + MINIF);
            }
        }
        __bf16 wh = (__bf16)w;
        W[bg * Vn + vg]  = wh;
        WT[vg * Bn + bg] = wh;
    }
    float qacc = qacc2 * LN2F;

    // deterministic block tree-reduction of qz
    red[tid] = qacc;
    __syncthreads();
    for (int s = 128; s > 0; s >>= 1) {
        if (tid < s) red[tid] += red[tid + s];
        __syncthreads();
    }
    if (tid == 0) qzPart[blockIdx.y * gridDim.x + blockIdx.x] = red[0];
}

// ---------------- Kernel 5: G = W . phi  -> temp_exp_m = theta*G ; new_exp_m -----------
// 64 waves total (16 m-tiles x 4 n-tiles), K loop over V.
__global__ __launch_bounds__(256)
void k_gemm_m(const __bf16* __restrict__ W, const __bf16* __restrict__ phiT,
              const float* __restrict__ thF, const float* __restrict__ expm,
              const int* __restrict__ bidx, const int* __restrict__ iter_n,
              float* __restrict__ out_tem, float* __restrict__ out_nem) {
    int tid = threadIdx.x, lane = tid & 31, wave = tid >> 5;
    int wid = blockIdx.x * 8 + wave;
    int mt = wid >> 2, nt = wid & 3;
    int n = lane & 15, hi = lane >> 4, khalf = hi * 8;
    v8f acc = {};
    for (int kc = 0; kc < Vn / 32; ++kc) {
        v16bf a, bmat;
        const __bf16* ap0 = W + (mt * 16 + n) * Vn + kc * 32 + hi * 16;
        const __bf16* ap1 = ap0 + 8;
#pragma unroll
        for (int e = 0; e < 8; ++e) { a[e] = ap0[e]; a[8 + e] = ap1[e]; }
        const __bf16* bp = phiT + (nt * 16 + n) * Vn + kc * 32 + hi * 16;
#pragma unroll
        for (int e = 0; e < 16; ++e) bmat[e] = bp[e];
        acc = __builtin_amdgcn_wmma_f32_16x16x32_bf16(false, a, false, bmat,
                                                      (short)0, acc, false, false);
    }
    float rho = fast_rho(*iter_n);
#pragma unroll
    for (int j = 0; j < 8; ++j) {
        int b = mt * 16 + j + 8 * hi;
        int k = nt * 16 + n;
        float tem = thF[b * Kn + k] * acc[j];
        out_tem[b * Kn + k] = tem;
        int d = bidx[b];
        out_nem[b * Kn + k] = (1.0f - rho) * expm[d * Kn + k] + rho * tem;
    }
}

// ---------------- Kernel 6: T = W^T . theta -> temp_exp_n = phi*T ; new_exp_n ----------
// 1024 waves total (256 m-tiles x 4 n-tiles), K loop over B.
__global__ __launch_bounds__(256)
void k_gemm_n(const __bf16* __restrict__ WT, const __bf16* __restrict__ thT,
              const float* __restrict__ phF, const float* __restrict__ expn,
              const int* __restrict__ iter_n, const int* __restrict__ C_m,
              const int* __restrict__ batch_C,
              float* __restrict__ out_ten, float* __restrict__ out_nen) {
    int tid = threadIdx.x, lane = tid & 31, wave = tid >> 5;
    int wid = blockIdx.x * 8 + wave;
    int mt = wid >> 2, nt = wid & 3;
    int n = lane & 15, hi = lane >> 4, khalf = hi * 8;
    v8f acc = {};
    for (int kc = 0; kc < Bn / 32; ++kc) {
        v16bf a, bmat;
        const __bf16* ap0 = WT + (mt * 16 + n) * Bn + kc * 32 + hi * 16;
        const __bf16* ap1 = ap0 + 8;
#pragma unroll
        for (int e = 0; e < 8; ++e) { a[e] = ap0[e]; a[8 + e] = ap1[e]; }
        const __bf16* bp = thT + (nt * 16 + n) * Bn + kc * 32 + hi * 16;
#pragma unroll
        for (int e = 0; e < 16; ++e) bmat[e] = bp[e];
        acc = __builtin_amdgcn_wmma_f32_16x16x32_bf16(false, a, false, bmat,
                                                      (short)0, acc, false, false);
    }
    float rho = fast_rho(*iter_n);
    float scale = (float)(*C_m) / (float)(*batch_C);
#pragma unroll
    for (int j = 0; j < 8; ++j) {
        int v = mt * 16 + j + 8 * hi;
        int k = nt * 16 + n;
        float ten = phF[v * Kn + k] * acc[j];
        out_ten[v * Kn + k] = ten;
        out_nen[v * Kn + k] = (1.0f - rho) * expn[v * Kn + k] + rho * ten * scale;
    }
}

// ---------------- Kernel 7: final deterministic qz reduction ---------------------------
__global__ __launch_bounds__(256)
void k_qz(const float* __restrict__ qzPart, float* __restrict__ out_qz) {
    __shared__ float red[256];
    int t = threadIdx.x;
    red[t] = qzPart[t] + qzPart[t + 256];   // 512 partials (32x16 blocks)
    __syncthreads();
    for (int s = 128; s > 0; s >>= 1) {
        if (t < s) red[t] += red[t + s];
        __syncthreads();
    }
    if (t == 0) *out_qz = red[0];
}

extern "C" void kernel_launch(void* const* d_in, const int* in_sizes, int n_in,
                              void* d_out, int out_size, void* d_ws, size_t ws_size,
                              hipStream_t stream) {
    const int*   bow   = (const int*)  d_in[0];
    const int*   bidx  = (const int*)  d_in[1];
    const float* alpha = (const float*)d_in[2];
    const float* pi    = (const float*)d_in[3];
    const float* expm  = (const float*)d_in[4];
    const float* beta  = (const float*)d_in[5];
    const float* expn  = (const float*)d_in[6];
    const int*   itn   = (const int*)  d_in[7];
    const int*   cm    = (const int*)  d_in[8];
    const int*   bc    = (const int*)  d_in[9];

    float* out      = (float*)d_out;
    float* out_ten  = out;              // temp_exp_n [V,K]
    float* out_tem  = out + 262144;     // temp_exp_m [B,K]
    float* out_qz   = out + 278528;     // exp_q_z scalar
    float* out_nen  = out + 278529;     // new_exp_n [V,K]
    float* out_nem  = out + 540673;     // new_exp_m_batch [B,K]

    char* ws = (char*)d_ws;
    float*  denom = (float*)(ws + 0);           //    64 f32
    float*  thF   = (float*)(ws + 256);         // 16384 f32
    float*  phF   = (float*)(ws + 65792);       // 262144 f32
    __bf16* thB   = (__bf16*)(ws + 1114368);    // 16384 bf16
    __bf16* thT   = (__bf16*)(ws + 1147136);    // 16384 bf16 (K-major)
    __bf16* phB   = (__bf16*)(ws + 1179904);    // 262144 bf16
    __bf16* phT   = (__bf16*)(ws + 1704192);    // 262144 bf16 (K-major)
    __bf16* W     = (__bf16*)(ws + 2228480);    // B*V bf16
    __bf16* WT    = (__bf16*)(ws + 4325632);    // V*B bf16
    float*  qzP   = (float*)(ws + 6422784);     // 512 f32 partials

    k_denom<<<1, 256, 0, stream>>>(beta, expn, denom);
    k_phi<<<(Vn * Kn) / 256, 256, 0, stream>>>(beta, expn, denom, phF, phB, phT);
    k_theta<<<(Bn * Kn) / 256, 256, 0, stream>>>(alpha, pi, expm, bidx, thF, thB, thT);
    dim3 g4(Vn / 128, Bn / 16);
    k_gamma<<<g4, 256, 0, stream>>>(bow, thF, phF, thB, phB, W, WT, qzP);
    k_gemm_m<<<8, 256, 0, stream>>>(W, phT, thF, expm, bidx, itn, out_tem, out_nem);
    k_gemm_n<<<128, 256, 0, stream>>>(WT, thT, phF, expn, itn, cm, bc, out_ten, out_nen);
    k_qz<<<1, 256, 0, stream>>>(qzP, out_qz);
}
